// PointTransformerLayer_6708738916629
// MI455X (gfx1250) — matrix operand order repaired and checked by reading, not compiled, not run on hardware
//
#include <hip/hip_runtime.h>
#include <hip/hip_bf16.h>

#define NB 2
#define NP 8192
#define NC 128
#define NK 16
#define BN_EPS 1e-5f

typedef __bf16 bf_t;
typedef __attribute__((ext_vector_type(16))) __bf16 v16bf;
typedef __attribute__((ext_vector_type(8)))  __bf16 v8bf;
typedef __attribute__((ext_vector_type(2)))  __bf16 v2bf;
typedef __attribute__((ext_vector_type(8)))  float  v8f;

__device__ __forceinline__ bf_t f2bf(float f) { return (bf_t)f; }

// ---------------------------------------------------------------------------
// Fragment-ordered storage.
//
// A-fragments (weights, prepared once in workspace as bf16):
//   flat = ((mt*4 + kt)*32 + lane)*16 + e
//   m = mt*16 + (lane&15)
//   k = kt*32 + (lane>>4)*16 + e        (contiguous k order, matches B swizzle)
//
// B tile in LDS, swizzled so each lane's fragment is contiguous:
//   element (k, n) -> Bsw[(k>>5)][n>>4][((k>>4)&1)*16 + (n&15)][k&15]
//   fragment (kt, nt) for lane: 16 contiguous bf16 (32 bytes) -> 2x ds_load_b128
// ---------------------------------------------------------------------------

__device__ __forceinline__ v16bf load_a_frag(const bf_t* Af, int mt, int kt, int lane) {
  return *(const v16bf*)(Af + ((mt * 4 + kt) * 32 + lane) * 16);
}

__device__ __forceinline__ v16bf load_b_frag(const bf_t* B, int kt, int nt, int lane) {
  return *(const v16bf*)(B + ((kt * 8 + nt) * 32 + lane) * 16);
}

// One wave computes D tiles (mt = wid, nt = 0..7) of a [128x128]x[128x128]
// GEMM, K=128: 32 x v_wmma_f32_16x16x32_bf16.
__device__ __forceinline__ void wave_gemm(const bf_t* Af, const bf_t* Blds,
                                          float acc[8][8], int wid, int lane) {
  v16bf afr[4];
#pragma unroll
  for (int kt = 0; kt < 4; ++kt) afr[kt] = load_a_frag(Af, wid, kt, lane);
#pragma unroll
  for (int nt = 0; nt < 8; ++nt) {
    v8f c = {0.f, 0.f, 0.f, 0.f, 0.f, 0.f, 0.f, 0.f};
#pragma unroll
    for (int kt = 0; kt < 4; ++kt) {
      v16bf bfr = load_b_frag(Blds, kt, nt, lane);
      c = __builtin_amdgcn_wmma_f32_16x16x32_bf16(false, afr[kt], false, bfr,
                                                  (short)0, c, false, false);
    }
#pragma unroll
    for (int r = 0; r < 8; ++r) acc[nt][r] = c[r];
  }
}

// ------------------------------------------------ weight prep: f32 -> bf16
__global__ __launch_bounds__(256) void prep_kernel(
    const float* __restrict__ wq, const float* __restrict__ wk,
    const float* __restrict__ wv, const float* __restrict__ w1,
    const float* __restrict__ w2, bf_t* __restrict__ out) {
  const int mat = blockIdx.x;
  const float* W = (mat == 0) ? wq : (mat == 1) ? wk : (mat == 2) ? wv
                   : (mat == 3) ? w1 : w2;
  bf_t* o = out + (size_t)mat * (NC * NC);
  for (int i = threadIdx.x; i < NC * NC; i += 256) {
    const int e = i & 15, lane = (i >> 4) & 31, kt = (i >> 9) & 3, mt = i >> 11;
    const int m = mt * 16 + (lane & 15);
    const int k = kt * 32 + (lane >> 4) * 16 + e;
    o[i] = f2bf(W[m * NC + k]);
  }
}

// ---------------------------------------------------------------- Q/K/V GEMM
__global__ __launch_bounds__(256) void qkv_kernel(
    const float* __restrict__ x,
    const float* __restrict__ bq, const float* __restrict__ bk,
    const float* __restrict__ bvp, const bf_t* __restrict__ wbf,
    float* __restrict__ qo, float* __restrict__ ko, float* __restrict__ vo) {
  __shared__ __align__(32) bf_t Bsw[4 * 8 * 32 * 16];   // 32KB swizzled B tile
  const int tid = threadIdx.x, lane = tid & 31, wid = tid >> 5;
  const int tpb = NP / 128;
  const int proj = blockIdx.x / (NB * tpb);
  const int rem  = blockIdx.x % (NB * tpb);
  const int bb = rem / tpb;
  const int n0 = (rem % tpb) * 128;
  const float* Bi = (proj == 0) ? bq : ((proj == 1) ? bk : bvp);
  float* Out      = (proj == 0) ? qo : ((proj == 1) ? ko : vo);
  const bf_t* Af  = wbf + (size_t)proj * (NC * NC);

  // Stage x tile: two channels per iteration -> packed 4B LDS stores.
  for (int i = tid; i < NC * 128 / 2; i += 256) {
    const int col = i & 127;
    const int c0 = (i >> 7) * 2;
    const float f0 = x[((size_t)bb * NC + c0) * NP + n0 + col];
    const float f1 = x[((size_t)bb * NC + c0 + 1) * NP + n0 + col];
    v2bf pk; pk[0] = f2bf(f0); pk[1] = f2bf(f1);
    bf_t* dst = Bsw + (((c0 >> 5) * 8 + (col >> 4)) * 32 +
                       ((c0 >> 4) & 1) * 16 + (col & 15)) * 16 + (c0 & 15);
    *(v2bf*)dst = pk;
  }
  __syncthreads();

  float acc[8][8];
  wave_gemm(Af, Bsw, acc, wid, lane);

  const int half = lane >> 4, ln = lane & 15;
  const int row0 = wid * 16 + 8 * half;
  const float4 bv0 = *(const float4*)(Bi + row0);
  const float4 bv1 = *(const float4*)(Bi + row0 + 4);
#pragma unroll
  for (int nt = 0; nt < 8; ++nt) {
    const int col = n0 + nt * 16 + ln;
    float* op = Out + ((size_t)bb * NP + col) * NC + row0;  // [b][pt][C]
    float4 o0 = make_float4(acc[nt][0] + bv0.x, acc[nt][1] + bv0.y,
                            acc[nt][2] + bv0.z, acc[nt][3] + bv0.w);
    float4 o1 = make_float4(acc[nt][4] + bv1.x, acc[nt][5] + bv1.y,
                            acc[nt][6] + bv1.z, acc[nt][7] + bv1.w);
    *(float4*)op = o0;
    *(float4*)(op + 4) = o1;
  }
}

// ------------------------------------------------------------------- 16-NN
__global__ __launch_bounds__(256) void knn_kernel(const float* __restrict__ p,
                                                  int* __restrict__ idx) {
  __shared__ float tile[256 * 3];
  const int tid = threadIdx.x;
  const int bpb = NP / 256;
  const int bb = blockIdx.x / bpb;
  const int i  = (blockIdx.x % bpb) * 256 + tid;
  const float px = p[((size_t)bb * NP + i) * 3 + 0];
  const float py = p[((size_t)bb * NP + i) * 3 + 1];
  const float pz = p[((size_t)bb * NP + i) * 3 + 2];
  float bd[NK]; int bi[NK];
#pragma unroll
  for (int s = 0; s < NK; ++s) { bd[s] = 3.4e38f; bi[s] = 0; }
  for (int jt = 0; jt < NP; jt += 256) {
    __syncthreads();
    for (int l = tid; l < 256 * 3; l += 256)
      tile[l] = p[((size_t)bb * NP + jt) * 3 + l];
    __syncthreads();
    for (int j2 = 0; j2 < 256; ++j2) {
      const float dx = tile[j2 * 3 + 0] - px;
      const float dy = tile[j2 * 3 + 1] - py;
      const float dz = tile[j2 * 3 + 2] - pz;
      const float d2 = dx * dx + dy * dy + dz * dz;
      if (d2 < bd[NK - 1]) {
        const int j = jt + j2;
#pragma unroll
        for (int s = NK - 1; s >= 0; --s) {
          const float pd = (s > 0) ? bd[s - 1] : -1.0f;
          const int   pi = (s > 0) ? bi[s - 1] : 0;
          if (d2 < pd)         { bd[s] = pd; bi[s] = pi; }
          else if (d2 < bd[s]) { bd[s] = d2; bi[s] = j;  }
        }
      }
    }
  }
#pragma unroll
  for (int s = 0; s < NK; ++s) idx[((size_t)bb * NP + i) * NK + s] = bi[s];
}

// -------------------------------------------- fused PE-MLP + attention MLP
__global__ __launch_bounds__(256) void attn_kernel(
    const float* __restrict__ pc,
    const float* __restrict__ qws, const float* __restrict__ kws,
    const float* __restrict__ vws, const int* __restrict__ idxg,
    const bf_t* __restrict__ wbf,
    const float* __restrict__ pe_w1,
    const float* __restrict__ pe_g, const float* __restrict__ pe_b,
    const float* __restrict__ pe_m, const float* __restrict__ pe_v,
    const float* __restrict__ pe_w2, const float* __restrict__ pe_b2,
    const float* __restrict__ bn1g, const float* __restrict__ bn1b,
    const float* __restrict__ bn1m, const float* __restrict__ bn1v,
    const float* __restrict__ bn2g, const float* __restrict__ bn2b,
    const float* __restrict__ bn2m, const float* __restrict__ bn2v,
    const float* __restrict__ atb2,
    float* __restrict__ out) {
  __shared__ __align__(32) bf_t Bsw[4 * 8 * 32 * 16];  // 32KB shared B tile
  __shared__ float  q2[8 * NC];        // bn1-prescaled q (+ folded offsets)
  __shared__ int    idx_lds[128];
  __shared__ float4 h4[128];           // (h0,h1,h2,1) per column
  __shared__ float4 sw2[NC];           // (s*w2_0, s*w2_1, s*w2_2, s)
  __shared__ float4 w2p[NC];           // (w2_0, w2_1, w2_2, pe_b2)
  __shared__ float  qoff[NC];          // bn1_shift + s*pe_b2
  __shared__ float  bn2s[NC], bn2o[NC], b2l[NC];

  const int tid = threadIdx.x, lane = tid & 31, wid = tid >> 5;
  const int half = lane >> 4, ln = lane & 15;
  const int bb = blockIdx.x / (NP / 8);
  const int p0 = (blockIdx.x % (NP / 8)) * 8;   // 8 points, 128 columns

  const bf_t* w1f = wbf + (size_t)3 * (NC * NC);
  const bf_t* w2f = wbf + (size_t)4 * (NC * NC);

  // Phase A1: indices + folded per-channel constants.
  if (tid < 128) {
    idx_lds[tid] = idxg[((size_t)bb * NP + p0) * NK + tid];
    const float s1 = bn1g[tid] * rsqrtf(bn1v[tid] + BN_EPS);
    const float o1 = bn1b[tid] - bn1m[tid] * s1;
    const float w20 = pe_w2[tid * 3 + 0], w21 = pe_w2[tid * 3 + 1],
                w22 = pe_w2[tid * 3 + 2], pb2 = pe_b2[tid];
    sw2[tid]  = make_float4(s1 * w20, s1 * w21, s1 * w22, s1);
    w2p[tid]  = make_float4(w20, w21, w22, pb2);
    qoff[tid] = o1 + s1 * pb2;
    const float s2 = bn2g[tid] * rsqrtf(bn2v[tid] + BN_EPS);
    bn2s[tid] = s2; bn2o[tid] = bn2b[tid] - bn2m[tid] * s2;
    b2l[tid]  = atb2[tid];
  }
  __syncthreads();

  // Phase A2: q2 = s1*q + (o1 + s1*pe_b2); Phase B: PE hidden per column.
  for (int i = tid; i < 8 * NC; i += 256) {
    const int c = i & 127;
    const float qv = qws[((size_t)bb * NP + p0 + (i >> 7)) * NC + c];
    q2[i] = fmaf(qv, sw2[c].w, qoff[c]);
  }
  if (tid < 128) {
    const int col = tid, pt = col >> 4;
    const int j = idx_lds[col];
    const float* pj = pc + ((size_t)bb * NP + j) * 3;
    const float* pi = pc + ((size_t)bb * NP + p0 + pt) * 3;
    const float rx = pj[0] - pi[0], ry = pj[1] - pi[1], rz = pj[2] - pi[2];
    float h[3];
#pragma unroll
    for (int o = 0; o < 3; ++o) {
      float t = pe_w1[o * 3 + 0] * rx + pe_w1[o * 3 + 1] * ry + pe_w1[o * 3 + 2] * rz;
      const float sc = pe_g[o] * rsqrtf(pe_v[o] + BN_EPS);
      t = (t - pe_m[o]) * sc + pe_b[o];
      h[o] = fmaxf(t, 0.f);
    }
    h4[col] = make_float4(h[0], h[1], h[2], 1.0f);
  }
  __syncthreads();

  // Phase C: B0 = relu(bn1(q - gk + n_r)); each thread builds 4 contiguous
  // 16-element fragment groups for one column -> packed 32B LDS stores.
  {
    const int col = tid & 127, pt = col >> 4;
    const int c0 = (tid >> 7) * 64;
    const int j = idx_lds[col];
    const float4 hc = h4[col];
    const float* kp = kws + ((size_t)bb * NP + j) * NC + c0;
    const float* qp = q2 + pt * NC + c0;
#pragma unroll
    for (int g = 0; g < 4; ++g) {
      const int cg = c0 + g * 16;
      const float4 k0 = *(const float4*)(kp + g * 16 + 0);
      const float4 k1 = *(const float4*)(kp + g * 16 + 4);
      const float4 k2 = *(const float4*)(kp + g * 16 + 8);
      const float4 k3 = *(const float4*)(kp + g * 16 + 12);
      const float kf[16] = {k0.x, k0.y, k0.z, k0.w, k1.x, k1.y, k1.z, k1.w,
                            k2.x, k2.y, k2.z, k2.w, k3.x, k3.y, k3.z, k3.w};
      const float4 q0 = *(const float4*)(qp + g * 16 + 0);
      const float4 q1 = *(const float4*)(qp + g * 16 + 4);
      const float4 q2v = *(const float4*)(qp + g * 16 + 8);
      const float4 q3 = *(const float4*)(qp + g * 16 + 12);
      const float qf[16] = {q0.x, q0.y, q0.z, q0.w, q1.x, q1.y, q1.z, q1.w,
                            q2v.x, q2v.y, q2v.z, q2v.w, q3.x, q3.y, q3.z, q3.w};
      v16bf pk;
#pragma unroll
      for (int e = 0; e < 16; ++e) {
        const float4 s = sw2[cg + e];
        const float val = qf[e] - s.w * kf[e] + s.x * hc.x + s.y * hc.y + s.z * hc.z;
        pk[e] = f2bf(fmaxf(val, 0.f));
      }
      bf_t* dst = Bsw + (((cg >> 5) * 8 + (col >> 4)) * 32 +
                         ((cg >> 4) & 1) * 16 + (col & 15)) * 16;
      *(v16bf*)dst = pk;
    }
  }
  __syncthreads();

  // Phase D: GEMM1 = at_w1 x B0
  float acc1[8][8];
  wave_gemm(w1f, Bsw, acc1, wid, lane);
  __syncthreads();

  // Phase E: B1 = relu(bn2(acc1)) -> packed 16B fragment-order stores.
  {
    const int r0 = wid * 16 + 8 * half;      // 8 consecutive rows per lane
    const float4 s0 = *(const float4*)(bn2s + r0);
    const float4 s1 = *(const float4*)(bn2s + r0 + 4);
    const float4 o0 = *(const float4*)(bn2o + r0);
    const float4 o1 = *(const float4*)(bn2o + r0 + 4);
    const float ss[8] = {s0.x, s0.y, s0.z, s0.w, s1.x, s1.y, s1.z, s1.w};
    const float oo[8] = {o0.x, o0.y, o0.z, o0.w, o1.x, o1.y, o1.z, o1.w};
    const int kt2 = wid >> 1;
#pragma unroll
    for (int nt = 0; nt < 8; ++nt) {
      v8bf pk;
#pragma unroll
      for (int r = 0; r < 8; ++r)
        pk[r] = f2bf(fmaxf(acc1[nt][r] * ss[r] + oo[r], 0.f));
      bf_t* dst = Bsw + ((kt2 * 8 + nt) * 32 + (wid & 1) * 16 + ln) * 16 + 8 * half;
      *(v8bf*)dst = pk;
    }
  }
  __syncthreads();

  // Phase F: GEMM2 = at_w2 x B1, + bias, softmax over the 16 neighbors
  // (16-lane shuffle reductions), weighted sum with n_v = gv + n_r.
  float acc2[8][8];
  wave_gemm(w2f, Bsw, acc2, wid, lane);

  const int r0 = wid * 16 + 8 * half;
  const float4 bb0 = *(const float4*)(b2l + r0);
  const float4 bb1 = *(const float4*)(b2l + r0 + 4);
  const float bias[8] = {bb0.x, bb0.y, bb0.z, bb0.w, bb1.x, bb1.y, bb1.z, bb1.w};
  float4 w2r[8];
#pragma unroll
  for (int r = 0; r < 8; ++r) w2r[r] = w2p[r0 + r];

#pragma unroll
  for (int nt = 0; nt < 8; ++nt) {
    const int pt = nt;
    const int col = pt * 16 + ln;
    const int j = idx_lds[col];
    const float* gvp = vws + ((size_t)bb * NP + j) * NC + r0;
    const float4 g0 = *(const float4*)gvp;
    const float4 g1 = *(const float4*)(gvp + 4);
    const float gv[8] = {g0.x, g0.y, g0.z, g0.w, g1.x, g1.y, g1.z, g1.w};
    const float4 hc = h4[col];
#pragma unroll
    for (int r = 0; r < 8; ++r) {
      const int row = r0 + r;
      const float av = acc2[nt][r] + bias[r];
      float mx = av;
      mx = fmaxf(mx, __shfl_xor(mx, 1));
      mx = fmaxf(mx, __shfl_xor(mx, 2));
      mx = fmaxf(mx, __shfl_xor(mx, 4));
      mx = fmaxf(mx, __shfl_xor(mx, 8));
      const float e = __expf(av - mx);
      float sum = e;
      sum += __shfl_xor(sum, 1);
      sum += __shfl_xor(sum, 2);
      sum += __shfl_xor(sum, 4);
      sum += __shfl_xor(sum, 8);
      const float wgt = e / sum;
      const float nr = w2r[r].x * hc.x + w2r[r].y * hc.y + w2r[r].z * hc.z + w2r[r].w;
      float t = wgt * (gv[r] + nr);
      t += __shfl_xor(t, 1);
      t += __shfl_xor(t, 2);
      t += __shfl_xor(t, 4);
      t += __shfl_xor(t, 8);
      if (ln == 0) out[((size_t)bb * NC + row) * NP + p0 + pt] = t;
    }
  }
}

extern "C" void kernel_launch(void* const* d_in, const int* in_sizes, int n_in,
                              void* d_out, int out_size, void* d_ws, size_t ws_size,
                              hipStream_t stream) {
  const float* p     = (const float*)d_in[0];
  const float* x     = (const float*)d_in[1];
  const float* wq    = (const float*)d_in[2];
  const float* bq    = (const float*)d_in[3];
  const float* wk    = (const float*)d_in[4];
  const float* bk    = (const float*)d_in[5];
  const float* wv    = (const float*)d_in[6];
  const float* bvp   = (const float*)d_in[7];
  const float* pe_w1 = (const float*)d_in[8];
  const float* pe_g  = (const float*)d_in[9];
  const float* pe_b  = (const float*)d_in[10];
  const float* pe_m  = (const float*)d_in[11];
  const float* pe_v  = (const float*)d_in[12];
  const float* pe_w2 = (const float*)d_in[13];
  const float* pe_b2 = (const float*)d_in[14];
  const float* bn1g  = (const float*)d_in[15];
  const float* bn1b  = (const float*)d_in[16];
  const float* bn1m  = (const float*)d_in[17];
  const float* bn1v  = (const float*)d_in[18];
  const float* w1    = (const float*)d_in[19];
  const float* bn2g  = (const float*)d_in[20];
  const float* bn2b  = (const float*)d_in[21];
  const float* bn2m  = (const float*)d_in[22];
  const float* bn2v  = (const float*)d_in[23];
  const float* w2    = (const float*)d_in[24];
  const float* atb2  = (const float*)d_in[25];
  float* out = (float*)d_out;

  const size_t BNC = (size_t)NB * NP * NC;
  float* qws = (float*)d_ws;                 // [b][n][C]
  float* kws = qws + BNC;                    // [b][n][C]
  float* vws = kws + BNC;                    // [b][n][C]
  int*   idxws = (int*)(vws + BNC);          // [b][n][16]
  bf_t*  wbf = (bf_t*)(idxws + (size_t)NB * NP * NK);  // 5 x 128x128 bf16 frags

  prep_kernel<<<5, 256, 0, stream>>>(wq, wk, wv, w1, w2, wbf);
  qkv_kernel<<<3 * NB * (NP / 128), 256, 0, stream>>>(x, bq, bk, bvp, wbf,
                                                      qws, kws, vws);
  knn_kernel<<<NB * (NP / 256), 256, 0, stream>>>(p, idxws);
  attn_kernel<<<NB * (NP / 8), 256, 0, stream>>>(p, qws, kws, vws, idxws, wbf,
      pe_w1, pe_g, pe_b, pe_m, pe_v, pe_w2, pe_b2,
      bn1g, bn1b, bn1m, bn1v, bn2g, bn2b, bn2m, bn2v, atb2, out);

  (void)in_sizes; (void)n_in; (void)out_size; (void)ws_size;
}
